// MultiHeadSelfAttention_71451075936355
// MI455X (gfx1250) — matrix operand order repaired and checked by reading, not compiled, not run on hardware
//
#include <hip/hip_runtime.h>
#include <hip/hip_bf16.h>

typedef __attribute__((ext_vector_type(16))) __bf16 v16bf;
typedef __attribute__((ext_vector_type(8)))  float  v8f;

#define D_MODEL 1024
#define NHEAD   16
#define SEQ     2048
#define NBATCH  4
#define DHEAD   64

#if __has_builtin(__builtin_amdgcn_global_load_async_to_lds_b128) && \
    __has_builtin(__builtin_amdgcn_s_wait_asynccnt)
#define HAS_ASYNC_LDS 1
typedef int v4i_vs __attribute__((vector_size(16)));
typedef __attribute__((address_space(1))) v4i_vs* gbl_b128_t;   // global int4*
typedef __attribute__((address_space(3))) v4i_vs* lds_b128_t;   // LDS int4*
#else
#define HAS_ASYNC_LDS 0
#endif

__device__ __forceinline__ unsigned short f2bf(float f) {
    unsigned int u = __float_as_uint(f);
    u += 0x7FFFu + ((u >> 16) & 1u);          // round-to-nearest-even
    return (unsigned short)(u >> 16);
}

__device__ __forceinline__ v16bf mkfrag(uint4 lo, uint4 hi) {
    union { uint4 u[2]; v16bf v; } c;
    c.u[0] = lo; c.u[1] = hi;
    return c.v;
}

__device__ __forceinline__ v8f wmma_bf16(v16bf a, v16bf b, v8f c) {
    return __builtin_amdgcn_wmma_f32_16x16x32_bf16(false, a, false, b,
                                                   (short)0, c, false, false);
}

// ---------------------------------------------------------------------------
// Kernel 1: fused Q/K/V projection (bf16 WMMA) + RoPE on Q,K.
//   q[m,e] = sum_k x[m,k] * wq[e,k]   (einsum 'bsd,ed->bse')
// Block tile: 128(M) x 64(N), 8 waves, each wave 16(M) x 64(N) for all of QKV.
// Outputs: Qws,Kws bf16 [b,h,s,dh];  Vt bf16 [b,h,dh,s] (transposed for PV).
// ---------------------------------------------------------------------------
__global__ __launch_bounds__(256)
void qkv_rope_kernel(const float* __restrict__ x,
                     const float* __restrict__ wq,
                     const float* __restrict__ wk,
                     const float* __restrict__ wv,
                     const int*   __restrict__ tpos,
                     unsigned short* __restrict__ Qws,
                     unsigned short* __restrict__ Kws,
                     unsigned short* __restrict__ Vt) {
    __shared__ alignas(16) unsigned short As[128 * 32];
    __shared__ alignas(16) unsigned short Bs[3][64 * 32];

    const int tid  = threadIdx.x;
    const int wid  = tid >> 5;
    const int lane = tid & 31;
    const int l16  = lane & 15;
    const int lh   = lane >> 4;           // 0: lanes 0-15, 1: lanes 16-31
    const int m0   = blockIdx.x * 128;
    const int n0   = blockIdx.y * 64;
    const float* wptr[3] = {wq, wk, wv};

    v8f acc[3][4];
    #pragma unroll
    for (int w = 0; w < 3; ++w)
        #pragma unroll
        for (int nt = 0; nt < 4; ++nt)
            #pragma unroll
            for (int r = 0; r < 8; ++r) acc[w][nt][r] = 0.f;

    for (int k0 = 0; k0 < D_MODEL; k0 += 32) {
        // ---- stage A tile (128x32 f32 -> bf16) ----
        #pragma unroll
        for (int i = 0; i < 4; ++i) {
            int c = tid + 256 * i;
            int row = c >> 3, c4 = c & 7;
            float4 v = *(const float4*)(x + (size_t)(m0 + row) * D_MODEL + k0 + c4 * 4);
            unsigned int lo = (unsigned)f2bf(v.x) | ((unsigned)f2bf(v.y) << 16);
            unsigned int hi = (unsigned)f2bf(v.z) | ((unsigned)f2bf(v.w) << 16);
            *(uint2*)(&As[row * 32 + c4 * 4]) = make_uint2(lo, hi);
        }
        // ---- stage 3 B tiles (64x32 each, [n][k] = weights' native layout) ----
        #pragma unroll
        for (int w = 0; w < 3; ++w)
            #pragma unroll
            for (int i = 0; i < 2; ++i) {
                int c = tid + 256 * i;
                int row = c >> 3, c4 = c & 7;
                float4 v = *(const float4*)(wptr[w] + (size_t)(n0 + row) * D_MODEL + k0 + c4 * 4);
                unsigned int lo = (unsigned)f2bf(v.x) | ((unsigned)f2bf(v.y) << 16);
                unsigned int hi = (unsigned)f2bf(v.z) | ((unsigned)f2bf(v.w) << 16);
                *(uint2*)(&Bs[w][row * 32 + c4 * 4]) = make_uint2(lo, hi);
            }
        // prefetch next k-tile (emits global_prefetch_b8)
        if (k0 + 32 < D_MODEL) {
            __builtin_prefetch(x + (size_t)(m0 + (tid >> 3)) * D_MODEL + (k0 + 32) + (tid & 7) * 4, 0, 0);
            __builtin_prefetch(wq + (size_t)(n0 + (tid >> 3)) * D_MODEL + (k0 + 32) + (tid & 7) * 4, 0, 0);
        }
        __syncthreads();

        // A fragment: lane<16 holds row=l16, K {0..7,16..23}; lane>=16: K {8..15,24..31}
        const unsigned short* ap = &As[(wid * 16 + l16) * 32 + lh * 8];
        v16bf af = mkfrag(*(const uint4*)ap, *(const uint4*)(ap + 16));

        #pragma unroll
        for (int w = 0; w < 3; ++w)
            #pragma unroll
            for (int nt = 0; nt < 4; ++nt) {
                // B fragment: lane holds N=l16, K {0..7,16..23} (lh=0) / {8..15,24..31} (lh=1)
                const unsigned short* bp = &Bs[w][(nt * 16 + l16) * 32 + lh * 8];
                v16bf bf = mkfrag(*(const uint4*)bp, *(const uint4*)(bp + 16));
                acc[w][nt] = wmma_bf16(af, bf, acc[w][nt]);
            }
        __syncthreads();
    }

    // ---- epilogue: RoPE on Q,K; store bf16 ----
    const int mrow0 = m0 + wid * 16 + lh * 8;
    float invf[4];
    #pragma unroll
    for (int nt = 0; nt < 4; ++nt) {
        int dh = (n0 + nt * 16 + l16) & 63;
        invf[nt] = __powf(10000.f, -(float)(dh & 62) * (1.f / 64.f));
    }
    #pragma unroll
    for (int r = 0; r < 8; ++r) {
        const int m = mrow0 + r;
        const int b = m >> 11, s = m & (SEQ - 1);
        const float pos = (float)tpos[s];
        #pragma unroll
        for (int nt = 0; nt < 4; ++nt) {
            const int e = n0 + nt * 16 + l16;
            const int h = e >> 6, dh = e & 63;
            float sn, cs;
            __sincosf(pos * invf[nt], &sn, &cs);
            float vq = acc[0][nt][r];
            float pq = __shfl_xor(vq, 1, 32);          // pair partner (e^1)
            float rq = (dh & 1) ? (pq * sn + vq * cs) : (vq * cs - pq * sn);
            float vk = acc[1][nt][r];
            float pk = __shfl_xor(vk, 1, 32);
            float rk = (dh & 1) ? (pk * sn + vk * cs) : (vk * cs - pk * sn);
            const size_t bh = (size_t)b * NHEAD + h;
            Qws[(bh * SEQ + s) * DHEAD + dh] = f2bf(rq);
            Kws[(bh * SEQ + s) * DHEAD + dh] = f2bf(rk);
            Vt[(bh * DHEAD + dh) * SEQ + s]  = f2bf(acc[2][nt][r]);
        }
    }
}

// ---------------------------------------------------------------------------
// Kernel 2: causal flash attention. Block = 128 q rows of one (b,h); 8 waves,
// each wave 16 q rows x Dh=64. K/V tiles (32 keys) staged block-wide into LDS
// via async global->LDS copies (ASYNCcnt) when available; all waves share them.
// scores = Q(16x64)*K^T (2 chained k=32 WMMAs per 16-key subtile); online
// softmax via shfl_xor row reductions; P re-laid-out via per-wave LDS slab;
// O += P(16x32)*V(32x64) (4 WMMAs).
// ---------------------------------------------------------------------------
__global__ __launch_bounds__(256)
void attn_kernel(const unsigned short* __restrict__ Qws,
                 const unsigned short* __restrict__ Kws,
                 const unsigned short* __restrict__ Vt,
                 float* __restrict__ attn) {
    __shared__ alignas(16) unsigned short KsT[32 * 64];     // [key][dh]
    __shared__ alignas(16) unsigned short VsT[64 * 32];     // [dh][key]
    __shared__ alignas(16) unsigned short Pl[8][16 * 32];   // per-wave P slab

    const int tid  = threadIdx.x;
    const int wid  = tid >> 5;
    const int lane = tid & 31;
    const int l16  = lane & 15;
    const int lh   = lane >> 4;
    const int bh   = blockIdx.x;                 // b*16 + h
    const int q0   = blockIdx.y * 128 + wid * 16;
    const int klo  = lh * 8;
    const float NEGINF = -__builtin_inff();

    // Q fragments (A-matrix), dh 0..31 and 32..63
    const unsigned short* qp = Qws + ((size_t)bh * SEQ + q0 + l16) * DHEAD;
    v16bf qf0 = mkfrag(*(const uint4*)(qp + klo),      *(const uint4*)(qp + klo + 16));
    v16bf qf1 = mkfrag(*(const uint4*)(qp + 32 + klo), *(const uint4*)(qp + 32 + klo + 16));

    v8f O[4];
    float mrow[8], lrow[8];
    #pragma unroll
    for (int nt = 0; nt < 4; ++nt)
        #pragma unroll
        for (int r = 0; r < 8; ++r) O[nt][r] = 0.f;
    #pragma unroll
    for (int r = 0; r < 8; ++r) { mrow[r] = NEGINF; lrow[r] = 0.f; }

    const int qbase = q0 + lh * 8;   // first row this lane's C slots cover

    // block-wide staging assignment (one b128 per thread per tile)
    const int krow = tid >> 3, kc = tid & 7;    // K: 32 rows x 8 x 16B
    const int vrow = tid >> 2, vc = tid & 3;    // V: 64 rows x 4 x 16B
    const unsigned short* gk = Kws + ((size_t)bh * SEQ + krow) * DHEAD + kc * 8;
    const unsigned short* gv = Vt  + ((size_t)bh * DHEAD + vrow) * SEQ + vc * 8;
    unsigned short* lk = &KsT[krow * 64 + kc * 8];
    unsigned short* lv = &VsT[vrow * 32 + vc * 8];

    const int kmax = blockIdx.y * 128 + 127;    // block-uniform key bound

    for (int k0 = 0; k0 <= kmax; k0 += 32) {
#if HAS_ASYNC_LDS
        __builtin_amdgcn_global_load_async_to_lds_b128(
            (gbl_b128_t)(gk + (size_t)k0 * DHEAD), (lds_b128_t)lk, 0, 0);
        __builtin_amdgcn_global_load_async_to_lds_b128(
            (gbl_b128_t)(gv + k0), (lds_b128_t)lv, 0, 0);
        __builtin_amdgcn_s_wait_asynccnt(0);
#else
        *(uint4*)lk = *(const uint4*)(gk + (size_t)k0 * DHEAD);
        *(uint4*)lv = *(const uint4*)(gv + k0);
#endif
        __syncthreads();

        if (k0 <= q0 + 15) {   // wave-uniform: EXEC stays all-ones inside
            float st[2][8];
            #pragma unroll
            for (int kt = 0; kt < 2; ++kt) {
                const int kloc = kt * 16 + l16;          // key within tile
                const int kpos = k0 + kloc;              // this lane's key col
                const unsigned short* kp = &KsT[kloc * 64 + lh * 8];
                v16bf kf0 = mkfrag(*(const uint4*)kp,        *(const uint4*)(kp + 16));
                v16bf kf1 = mkfrag(*(const uint4*)(kp + 32), *(const uint4*)(kp + 48));
                v8f sc;
                #pragma unroll
                for (int r = 0; r < 8; ++r) sc[r] = 0.f;
                sc = wmma_bf16(qf0, kf0, sc);
                sc = wmma_bf16(qf1, kf1, sc);
                #pragma unroll
                for (int r = 0; r < 8; ++r)
                    st[kt][r] = (kpos <= qbase + r) ? sc[r] * 0.125f : NEGINF; // 1/sqrt(64)
            }

            // online softmax per row (rows 0-7 in lanes 0-15, 8-15 in 16-31)
            #pragma unroll
            for (int r = 0; r < 8; ++r) {
                float v = fmaxf(st[0][r], st[1][r]);
                v = fmaxf(v, __shfl_xor(v, 1, 32));
                v = fmaxf(v, __shfl_xor(v, 2, 32));
                v = fmaxf(v, __shfl_xor(v, 4, 32));
                v = fmaxf(v, __shfl_xor(v, 8, 32));
                const float mn = fmaxf(mrow[r], v);
                const float a  = __expf(mrow[r] - mn);
                const float p0 = __expf(st[0][r] - mn);
                const float p1 = __expf(st[1][r] - mn);
                float rs = p0 + p1;
                rs += __shfl_xor(rs, 1, 32);
                rs += __shfl_xor(rs, 2, 32);
                rs += __shfl_xor(rs, 4, 32);
                rs += __shfl_xor(rs, 8, 32);
                lrow[r] = lrow[r] * a + rs;
                mrow[r] = mn;
                #pragma unroll
                for (int nt = 0; nt < 4; ++nt) O[nt][r] *= a;
                // C-layout -> row-major P tile in per-wave LDS slab
                Pl[wid][(r + lh * 8) * 32 + l16]      = f2bf(p0);
                Pl[wid][(r + lh * 8) * 32 + 16 + l16] = f2bf(p1);
            }

            // P as A-fragment (same-wave LDS ops are in order)
            const unsigned short* pp = &Pl[wid][l16 * 32 + klo];
            v16bf pf = mkfrag(*(const uint4*)pp, *(const uint4*)(pp + 16));

            #pragma unroll
            for (int nt = 0; nt < 4; ++nt) {
                // V B-fragment: N = dh (=nt*16+l16), K = key contiguous in VsT
                const unsigned short* vp = &VsT[(nt * 16 + l16) * 32 + lh * 8];
                v16bf vf = mkfrag(*(const uint4*)vp, *(const uint4*)(vp + 16));
                O[nt] = wmma_bf16(pf, vf, O[nt]);
            }
        }
        __syncthreads();
    }

    // normalize and store fp32 [b, s, h*64+dh]
    const int b = bh >> 4;
    const int h = bh & 15;
    #pragma unroll
    for (int r = 0; r < 8; ++r) {
        const int s = qbase + r;
        const float inv_l = 1.f / lrow[r];
        #pragma unroll
        for (int nt = 0; nt < 4; ++nt) {
            const int e = h * DHEAD + nt * 16 + l16;
            attn[((size_t)b * SEQ + s) * D_MODEL + e] = O[nt][r] * inv_l;
        }
    }
}

// ---------------------------------------------------------------------------
// Kernel 3: output projection  out[m,e] = sum_k attn[m,k] * wo[e,k]
// ---------------------------------------------------------------------------
__global__ __launch_bounds__(256)
void out_proj_kernel(const float* __restrict__ a_in,
                     const float* __restrict__ wo,
                     float* __restrict__ out) {
    __shared__ alignas(16) unsigned short As[128 * 32];
    __shared__ alignas(16) unsigned short Bs[64 * 32];

    const int tid  = threadIdx.x;
    const int wid  = tid >> 5;
    const int lane = tid & 31;
    const int l16  = lane & 15;
    const int lh   = lane >> 4;
    const int m0   = blockIdx.x * 128;
    const int n0   = blockIdx.y * 64;

    v8f acc[4];
    #pragma unroll
    for (int nt = 0; nt < 4; ++nt)
        #pragma unroll
        for (int r = 0; r < 8; ++r) acc[nt][r] = 0.f;

    for (int k0 = 0; k0 < D_MODEL; k0 += 32) {
        #pragma unroll
        for (int i = 0; i < 4; ++i) {
            int c = tid + 256 * i;
            int row = c >> 3, c4 = c & 7;
            float4 v = *(const float4*)(a_in + (size_t)(m0 + row) * D_MODEL + k0 + c4 * 4);
            unsigned int lo = (unsigned)f2bf(v.x) | ((unsigned)f2bf(v.y) << 16);
            unsigned int hi = (unsigned)f2bf(v.z) | ((unsigned)f2bf(v.w) << 16);
            *(uint2*)(&As[row * 32 + c4 * 4]) = make_uint2(lo, hi);
        }
        #pragma unroll
        for (int i = 0; i < 2; ++i) {
            int c = tid + 256 * i;
            int row = c >> 3, c4 = c & 7;
            float4 v = *(const float4*)(wo + (size_t)(n0 + row) * D_MODEL + k0 + c4 * 4);
            unsigned int lo = (unsigned)f2bf(v.x) | ((unsigned)f2bf(v.y) << 16);
            unsigned int hi = (unsigned)f2bf(v.z) | ((unsigned)f2bf(v.w) << 16);
            *(uint2*)(&Bs[row * 32 + c4 * 4]) = make_uint2(lo, hi);
        }
        if (k0 + 32 < D_MODEL) {
            __builtin_prefetch(a_in + (size_t)(m0 + (tid >> 3)) * D_MODEL + (k0 + 32) + (tid & 7) * 4, 0, 0);
            __builtin_prefetch(wo + (size_t)(n0 + (tid >> 3)) * D_MODEL + (k0 + 32) + (tid & 7) * 4, 0, 0);
        }
        __syncthreads();

        const unsigned short* ap = &As[(wid * 16 + l16) * 32 + lh * 8];
        v16bf af = mkfrag(*(const uint4*)ap, *(const uint4*)(ap + 16));
        #pragma unroll
        for (int nt = 0; nt < 4; ++nt) {
            const unsigned short* bp = &Bs[(nt * 16 + l16) * 32 + lh * 8];
            v16bf bf = mkfrag(*(const uint4*)bp, *(const uint4*)(bp + 16));
            acc[nt] = wmma_bf16(af, bf, acc[nt]);
        }
        __syncthreads();
    }

    const int mrow0 = m0 + wid * 16 + lh * 8;
    #pragma unroll
    for (int r = 0; r < 8; ++r)
        #pragma unroll
        for (int nt = 0; nt < 4; ++nt)
            out[(size_t)(mrow0 + r) * D_MODEL + n0 + nt * 16 + l16] = acc[nt][r];
}

// ---------------------------------------------------------------------------
extern "C" void kernel_launch(void* const* d_in, const int* in_sizes, int n_in,
                              void* d_out, int out_size, void* d_ws, size_t ws_size,
                              hipStream_t stream) {
    (void)in_sizes; (void)n_in; (void)out_size; (void)ws_size;
    const float* x  = (const float*)d_in[0];
    const float* wq = (const float*)d_in[1];
    const float* wk = (const float*)d_in[2];
    const float* wv = (const float*)d_in[3];
    const float* wo = (const float*)d_in[4];
    const int*   tp = (const int*)d_in[5];

    char* ws = (char*)d_ws;
    unsigned short* Qws = (unsigned short*)(ws);                              // 16 MB
    unsigned short* Kws = (unsigned short*)(ws + (size_t)16 * 1024 * 1024);   // 16 MB
    unsigned short* Vt  = (unsigned short*)(ws + (size_t)32 * 1024 * 1024);   // 16 MB
    float*          at  = (float*)        (ws + (size_t)48 * 1024 * 1024);    // 32 MB

    qkv_rope_kernel<<<dim3(64, 16), dim3(256), 0, stream>>>(x, wq, wk, wv, tp, Qws, Kws, Vt);
    attn_kernel    <<<dim3(64, 16), dim3(256), 0, stream>>>(Qws, Kws, Vt, at);
    out_proj_kernel<<<dim3(64, 16), dim3(256), 0, stream>>>(at, wo, (float*)d_out);
}
